// SaSamplingModule_38826504356627
// MI455X (gfx1250) — matrix operand order repaired and checked by reading, not compiled, not run on hardware
//
#include <hip/hip_runtime.h>
#include <hip/hip_bf16.h>
#include <stdint.h>

typedef __attribute__((ext_vector_type(16))) _Float16 v16h;
typedef __attribute__((ext_vector_type(8)))  _Float16 v8h;
typedef __attribute__((ext_vector_type(8)))  float    v8f;

#define BATCH   16
#define NPTS    16384
#define NCH     288
#define NPOINT  256
#define NSAMPLE 16
#define RAD2    0.09f
#define INV_RAD (1.0f/0.3f)
#define CIN0    291
#define KP0     320   // padded K for layer 0
#define KP12    288
#define NCOL    (BATCH*NPOINT*NSAMPLE)   // 65536

// ---------------------------------------------------------------------------
// 1) Farthest point sampling: one block per batch, points resident in VGPRs,
//    two-level (shuffle + LDS) argmax reduction per iteration.
// ---------------------------------------------------------------------------
#define FPS_T 1024
#define PPT   (NPTS/FPS_T)   // 16 points per thread

__global__ __launch_bounds__(FPS_T) void fps_kernel(
    const float* __restrict__ xyz, float* __restrict__ out_newxyz,
    int* __restrict__ out_inds)
{
  const int b = blockIdx.x;
  const int t = threadIdx.x;
  const float* X = xyz + (size_t)b * NPTS * 3;

  float px[PPT], py[PPT], pz[PPT], dist[PPT];
#pragma unroll
  for (int i = 0; i < PPT; ++i) {
    const int p = i * FPS_T + t;
    px[i] = X[p*3+0]; py[i] = X[p*3+1]; pz[i] = X[p*3+2];
    dist[i] = 1e10f;
  }

  __shared__ float sV[32];
  __shared__ int   sI[32];
  __shared__ int   sCur;

  const int lane = t & 31, wave = t >> 5;
  int curIdx = 0;

  for (int it = 0; it < NPOINT; ++it) {
    if (t == 0) {
      out_inds[b*NPOINT + it] = curIdx;
      out_newxyz[((size_t)b*NPOINT + it)*3 + 0] = X[curIdx*3+0];
      out_newxyz[((size_t)b*NPOINT + it)*3 + 1] = X[curIdx*3+1];
      out_newxyz[((size_t)b*NPOINT + it)*3 + 2] = X[curIdx*3+2];
    }
    const float cx = X[curIdx*3+0], cy = X[curIdx*3+1], cz = X[curIdx*3+2];

    float bestV = -1.0f; int bestI = 0;
#pragma unroll
    for (int i = 0; i < PPT; ++i) {
      const float dx = px[i]-cx, dy = py[i]-cy, dz = pz[i]-cz;
      const float d = dx*dx + dy*dy + dz*dz;
      dist[i] = fminf(dist[i], d);
      const int p = i * FPS_T + t;
      if (dist[i] > bestV || (dist[i] == bestV && p < bestI)) { bestV = dist[i]; bestI = p; }
    }
#pragma unroll
    for (int off = 16; off >= 1; off >>= 1) {
      const float ov = __shfl_down(bestV, off);
      const int   oi = __shfl_down(bestI, off);
      if (ov > bestV || (ov == bestV && oi < bestI)) { bestV = ov; bestI = oi; }
    }
    if (lane == 0) { sV[wave] = bestV; sI[wave] = bestI; }
    __syncthreads();
    if (wave == 0) {
      float v = sV[lane]; int ii = sI[lane];
#pragma unroll
      for (int off = 16; off >= 1; off >>= 1) {
        const float ov = __shfl_down(v, off);
        const int   oi = __shfl_down(ii, off);
        if (ov > v || (ov == v && oi < ii)) { v = ov; ii = oi; }
      }
      if (lane == 0) sCur = ii;
    }
    __syncthreads();
    curIdx = sCur;
  }
}

// ---------------------------------------------------------------------------
// 2) Ball query: thread per (b,s); 256 threads of a block stream the same
//    batch's xyz (L2-hot broadcast). First-16-in-index-order semantics.
// ---------------------------------------------------------------------------
__global__ __launch_bounds__(NPOINT) void ballq_kernel(
    const float* __restrict__ xyz, const float* __restrict__ newxyz,
    int* __restrict__ idxout)
{
  const int b = blockIdx.x, s = threadIdx.x;
  const float* X = xyz + (size_t)b * NPTS * 3;
  const int bs = b*NPOINT + s;
  const float qx = newxyz[bs*3+0], qy = newxyz[bs*3+1], qz = newxyz[bs*3+2];
  int* out = idxout + (size_t)bs * NSAMPLE;
  int cnt = 0, first = NPTS;
  for (int j = 0; j < NPTS && cnt < NSAMPLE; ++j) {
    const float dx = X[j*3+0]-qx, dy = X[j*3+1]-qy, dz = X[j*3+2]-qz;
    if (dx*dx + dy*dy + dz*dz < RAD2) {
      if (cnt == 0) first = j;
      out[cnt++] = j;
    }
  }
  for (; cnt < NSAMPLE; ++cnt) out[cnt] = first;
}

// ---------------------------------------------------------------------------
// 3) Gather + layer-0 input build: H0 column-major f16 [col][KP0],
//    rows 0..2 = (xyz - q)/R, 3..290 = features, 291..319 = zero pad.
// ---------------------------------------------------------------------------
__global__ __launch_bounds__(128) void gather_kernel(
    const float* __restrict__ xyz, const float* __restrict__ feat,
    const float* __restrict__ newxyz, const int* __restrict__ idx,
    _Float16* __restrict__ h0)
{
  const int bs = blockIdx.x;           // (b*NPOINT + s)
  const int b = bs >> 8;
  const int t = threadIdx.x, w = t >> 5, lane = t & 31;
  const float qx = newxyz[bs*3+0], qy = newxyz[bs*3+1], qz = newxyz[bs*3+2];
  for (int kk = w; kk < NSAMPLE; kk += 4) {
    const int j = idx[(size_t)bs*NSAMPLE + kk];
    _Float16* dst = h0 + ((size_t)bs*NSAMPLE + kk) * KP0;
    const float* P = xyz + ((size_t)b*NPTS + j) * 3;
    const float* F = feat + (size_t)b*NCH*NPTS + j;
    for (int r = lane; r < KP0; r += 32) {
      float v;
      if (r < 3)        v = (P[r] - (r == 0 ? qx : (r == 1 ? qy : qz))) * INV_RAD;
      else if (r < CIN0) v = F[(size_t)(r-3) * NPTS];
      else               v = 0.0f;
      dst[r] = (_Float16)v;
    }
  }
}

// ---------------------------------------------------------------------------
// 4) Pack weights to f16 (K-padded) + fold BN into per-channel scale/shift.
// ---------------------------------------------------------------------------
__global__ void prep_kernel(
    const float* __restrict__ W0, const float* __restrict__ W1, const float* __restrict__ W2,
    const float* __restrict__ g0, const float* __restrict__ bb0, const float* __restrict__ m0, const float* __restrict__ v0,
    const float* __restrict__ g1, const float* __restrict__ bb1, const float* __restrict__ m1, const float* __restrict__ v1,
    const float* __restrict__ g2, const float* __restrict__ bb2, const float* __restrict__ m2, const float* __restrict__ v2,
    _Float16* __restrict__ wf, float* __restrict__ ss)
{
  const int tid = blockIdx.x * blockDim.x + threadIdx.x;
  if (tid < NCH*KP0) {
    const int o = tid / KP0, k = tid % KP0;
    wf[tid] = (k < CIN0) ? (_Float16)W0[o*CIN0 + k] : (_Float16)0.0f;
  }
  if (tid < NCH*KP12) {
    wf[NCH*KP0 + tid]             = (_Float16)W1[tid];
    wf[NCH*KP0 + NCH*KP12 + tid]  = (_Float16)W2[tid];
  }
  if (tid < NCH) {
    const float s0 = g0[tid] * rsqrtf(v0[tid] + 1e-5f);
    ss[tid] = s0;            ss[NCH + tid] = bb0[tid] - m0[tid]*s0;
    const float s1 = g1[tid] * rsqrtf(v1[tid] + 1e-5f);
    ss[576 + tid] = s1;      ss[576 + NCH + tid] = bb1[tid] - m1[tid]*s1;
    const float s2 = g2[tid] * rsqrtf(v2[tid] + 1e-5f);
    ss[1152 + tid] = s2;     ss[1152 + NCH + tid] = bb2[tid] - m2[tid]*s2;
  }
}

// ---------------------------------------------------------------------------
// 5) WMMA GEMM + fused BN/ReLU epilogue, with the 96xKp weight panel staged
//    once per block into LDS via async global->LDS (ASYNCcnt) in a
//    chunk-swizzled layout: 16B chunk (row r, kchunk c) lives at
//    sA + (c*96 + r)*8 halves. Inner-loop ds_load_b128 reads are then
//    conflict-free and match the 16-bit WMMA A-operand VGPR layout.
//    Block = 1 m-group (96 rows) x 8 n-tiles; wave = 6 M-tiles x 1 N-tile.
// ---------------------------------------------------------------------------
__global__ __launch_bounds__(256) void gemm_kernel(
    const _Float16* __restrict__ A,    // [288][Kp] f16, row-major
    const _Float16* __restrict__ Bm,   // [NCOL][Kp] f16, column-major H
    const float* __restrict__ ss,      // scale[288], shift[288]
    _Float16* __restrict__ Hout,       // mode 0 output: [NCOL][288] f16
    float* __restrict__ Fout,          // mode 1 output (B,288,256) f32
    int Kp, int mode)
{
  __shared__ __align__(16) _Float16 sA[96 * KP0];   // 60 KB (Kp<=320)

  const int tid  = threadIdx.x;
  const int lane = tid & 31;
  const int wid  = tid >> 5;
  const int mg   = blockIdx.x % 3;     // m-group: rows [mg*96, mg*96+96)
  const int ng   = blockIdx.x / 3;     // 8 n-tiles per block
  const int nt   = ng*8 + wid;         // 0..4095
  const int ln   = lane & 15;
  const int hi   = lane >> 4;          // half-wave id
  const int col  = nt*16 + ln;
  const int mbase = mg * 96;

  // ---- async fill of the weight panel into LDS (chunk-swizzled) ----
  {
    const int kchunks = Kp >> 3;             // 16B chunks per row (40 / 36)
    const int total   = 96 * kchunks;
    for (int q = tid; q < total; q += 256) {
      const int r = q / kchunks;
      const int c = q - r * kchunks;
      const _Float16* src = A + (size_t)(mbase + r) * Kp + (c << 3);
      const uint32_t  dst = (uint32_t)(uintptr_t)(sA + ((c * 96 + r) << 3));
      asm volatile("global_load_async_to_lds_b128 %0, %1, off"
                   :: "v"(dst), "v"(src) : "memory");
    }
    asm volatile("s_wait_asynccnt 0" ::: "memory");
  }
  __syncthreads();

  const _Float16* bp = Bm + (size_t)col * Kp + hi*8;

  v8f c[6] = {};
  const int ksteps = Kp >> 5;
  for (int kt = 0; kt < ksteps; ++kt) {
    const int kb = kt * 32;
    // --- B operand: k chunks (kb/8+hi) and (kb/8+2+hi) of this lane's column (global) ---
    const v8h x0 = *(const v8h*)(bp + kb);
    const v8h x1 = *(const v8h*)(bp + kb + 16);
    v16h bt;
#pragma unroll
    for (int i = 0; i < 8; ++i) { bt[i] = x0[i]; bt[8+i] = x1[i]; }
    // --- 6 A tiles from LDS: chunks (kb/8+hi) and (kb/8+2+hi) of row ---
    const int c0 = (kb >> 3) + hi;
#pragma unroll
    for (int sub = 0; sub < 6; ++sub) {
      const int row = sub*16 + ln;
      const v8h a0 = *(const v8h*)(sA + ((c0       * 96 + row) << 3));
      const v8h a1 = *(const v8h*)(sA + (((c0 + 2) * 96 + row) << 3));
      v16h at;
#pragma unroll
      for (int i = 0; i < 8; ++i) { at[i] = a0[i]; at[8+i] = a1[i]; }
      c[sub] = __builtin_amdgcn_wmma_f32_16x16x32_f16(
          false, at, false, bt, (short)0, c[sub], false, false);
    }
  }

  // --- fused BN + ReLU epilogue ---
#pragma unroll
  for (int sub = 0; sub < 6; ++sub) {
    const int m0 = mbase + sub*16 + hi*8;   // this lane's 8 consecutive rows
    if (mode == 0) {
      v8h o;
#pragma unroll
      for (int r = 0; r < 8; ++r) {
        const float e = fmaxf(c[sub][r] * ss[m0+r] + ss[NCH + m0+r], 0.0f);
        o[r] = (_Float16)e;
      }
      *(v8h*)(Hout + (size_t)col * KP12 + m0) = o;
    } else {
      float e[8];
#pragma unroll
      for (int r = 0; r < 8; ++r)
        e[r] = fmaxf(c[sub][r] * ss[m0+r] + ss[NCH + m0+r], 0.0f);
      // max over the 16 columns of this (b,s) group: reduce across 16-lane half
#pragma unroll
      for (int off = 8; off >= 1; off >>= 1)
#pragma unroll
        for (int r = 0; r < 8; ++r)
          e[r] = fmaxf(e[r], __shfl_xor(e[r], off, 16));
      if (ln == 0) {
        const int b = nt >> 8, s = nt & 255;
#pragma unroll
        for (int r = 0; r < 8; ++r)
          Fout[((size_t)b*NCH + m0 + r) * NPOINT + s] = e[r];
      }
    }
  }
}

// ---------------------------------------------------------------------------
// Launch
// ---------------------------------------------------------------------------
extern "C" void kernel_launch(void* const* d_in, const int* in_sizes, int n_in,
                              void* d_out, int out_size, void* d_ws, size_t ws_size,
                              hipStream_t stream) {
  const float* xyz  = (const float*)d_in[0];
  const float* feat = (const float*)d_in[1];
  const float* W0 = (const float*)d_in[2];
  const float* g0 = (const float*)d_in[3];
  const float* b0 = (const float*)d_in[4];
  const float* m0 = (const float*)d_in[5];
  const float* v0 = (const float*)d_in[6];
  const float* W1 = (const float*)d_in[7];
  const float* g1 = (const float*)d_in[8];
  const float* b1 = (const float*)d_in[9];
  const float* m1 = (const float*)d_in[10];
  const float* v1 = (const float*)d_in[11];
  const float* W2 = (const float*)d_in[12];
  const float* g2 = (const float*)d_in[13];
  const float* b2 = (const float*)d_in[14];
  const float* m2 = (const float*)d_in[15];
  const float* v2 = (const float*)d_in[16];

  // d_out layout: new_xyz (B*256*3) | new_features (B*288*256) | inds (B*256)
  float* out_newxyz = (float*)d_out;
  float* out_feat   = (float*)d_out + (size_t)BATCH*NPOINT*3;
  int*   out_inds   = (int*)((float*)d_out + (size_t)BATCH*NPOINT*3
                                           + (size_t)BATCH*NCH*NPOINT);

  // workspace carve-up
  char* ws = (char*)d_ws;
  _Float16* h0  = (_Float16*)(ws);                                 // 65536*320*2 = 40 MiB
  _Float16* h1  = (_Float16*)(ws + (size_t)NCOL*KP0*2);            // 65536*288*2 = 36 MiB
  _Float16* wf  = (_Float16*)(ws + (size_t)NCOL*KP0*2 + (size_t)NCOL*KP12*2);
  float*    ss  = (float*)((char*)wf + ((size_t)NCH*KP0 + 2*(size_t)NCH*KP12)*2);
  int*      idx = (int*)((char*)ss + 3*576*sizeof(float));

  _Float16* wf0 = wf;
  _Float16* wf1 = wf + (size_t)NCH*KP0;
  _Float16* wf2 = wf + (size_t)NCH*KP0 + (size_t)NCH*KP12;

  // 1) FPS (writes new_xyz + inds straight into d_out)
  fps_kernel<<<BATCH, FPS_T, 0, stream>>>(xyz, out_newxyz, out_inds);
  // 2) ball query
  ballq_kernel<<<BATCH, NPOINT, 0, stream>>>(xyz, out_newxyz, idx);
  // 3) gather + f16 H0 build
  gather_kernel<<<BATCH*NPOINT, 128, 0, stream>>>(xyz, feat, out_newxyz, idx, h0);
  // 4) weight pack + BN fold
  prep_kernel<<<(NCH*KP0 + 255)/256, 256, 0, stream>>>(
      W0, W1, W2, g0, b0, m0, v0, g1, b1, m1, v1, g2, b2, m2, v2, wf, ss);
  // 5-7) three WMMA GEMM layers; layer 2 fuses the NSAMPLE max-pool
  const int gblocks = 3 * (NCOL / 16 / 8);   // 3 m-groups x 512 n-groups = 1536
  gemm_kernel<<<gblocks, 256, 0, stream>>>(wf0, h0, ss,        h1, nullptr, KP0,  0);
  gemm_kernel<<<gblocks, 256, 0, stream>>>(wf1, h1, ss + 576,  h0, nullptr, KP12, 0);
  gemm_kernel<<<gblocks, 256, 0, stream>>>(wf2, h0, ss + 1152, nullptr, out_feat, KP12, 1);
}
